// CrossAttention_60129542144729
// MI455X (gfx1250) — matrix operand order repaired and checked by reading, not compiled, not run on hardware
//
#include <hip/hip_runtime.h>
#include <hip/hip_bf16.h>

typedef unsigned short u16;
typedef __attribute__((ext_vector_type(16))) __bf16 v16bf;
typedef __attribute__((ext_vector_type(8)))  float  v8f;

union FragAB { v16bf v; uint4 u[2]; };

__device__ __forceinline__ v8f wmma_bf16(v16bf a, v16bf b, v8f c) {
    return __builtin_amdgcn_wmma_f32_16x16x32_bf16(false, a, false, b, (short)0, c, false, false);
}

__device__ __forceinline__ u16 f2bf(float f) {          // round-to-nearest-even fp32 -> bf16
    unsigned u = __float_as_uint(f);
    unsigned r = ((u >> 16) & 1u) + 0x7fffu;
    return (u16)((u + r) >> 16);
}

// sinusoidal positional encoding value: pe[pos][c], d_model = 512
__device__ __forceinline__ float pe_val(int pos, int c) {
    int i = c >> 1;
    float div = __expf(-(float)(2 * i) * (9.210340371976184f / 512.0f)); // ln(10000)/512
    float a = (float)pos * div;
    return (c & 1) ? __cosf(a) : __sinf(a);
}

#define B_SZ   8
#define C_SZ   512
#define L_SZ   4096
#define S_SZ   196
#define SPAD   224
#define HEADS  8
#define HD     64
#define PDIM   18                 // 16 + 1-voxel halo each side
#define PVOL   (PDIM*PDIM*PDIM)   // 5832 padded voxels per batch

// ---------------------------------------------------------------- utilities
__global__ void fill_zero_u16(u16* p, int n) {
    int i = blockIdx.x * 256 + threadIdx.x;
    if (i < n) p[i] = 0;
}

__global__ void cvt_bf16(const float* __restrict__ s, u16* __restrict__ d, int n) {
    int i = blockIdx.x * 256 + threadIdx.x;
    if (i < n) d[i] = f2bf(s[i]);
}

// conv_w [cout][cin][27] f32 -> wt [tap][cout][cin] bf16
__global__ void permute_convw(const float* __restrict__ w, u16* __restrict__ wt) {
    int i = blockIdx.x * 256 + threadIdx.x;
    if (i >= 512 * 512 * 27) return;
    int ci = i & 511, co = (i >> 9) & 511, tap = i >> 18;
    wt[i] = f2bf(w[((size_t)co * 512 + ci) * 27 + tap]);
}

// ---------------------------------------------------------------- GroupNorm
__global__ void gn_stats(const float* __restrict__ x, float* __restrict__ stats) {
    __shared__ float ss[256], sq[256];
    int bg = blockIdx.x;                          // b*32 + g ; 16 ch * 4096 vox contiguous
    const float* base = x + (size_t)bg * 65536;
    float s = 0.f, q = 0.f;
    for (int i = threadIdx.x; i < 65536; i += 256) { float v = base[i]; s += v; q += v * v; }
    ss[threadIdx.x] = s; sq[threadIdx.x] = q;
    __syncthreads();
    for (int st = 128; st > 0; st >>= 1) {
        if (threadIdx.x < st) { ss[threadIdx.x] += ss[threadIdx.x + st]; sq[threadIdx.x] += sq[threadIdx.x + st]; }
        __syncthreads();
    }
    if (threadIdx.x == 0) {
        float mean = ss[0] * (1.f / 65536.f);
        float var  = sq[0] * (1.f / 65536.f) - mean * mean;
        stats[bg * 2] = mean;
        stats[bg * 2 + 1] = rsqrtf(var + 1e-5f);
    }
}

// normalize + scale/bias + SiLU, write bf16 into halo-padded channels-last h[b][18][18][18][c]
__global__ void gn_apply(const float* __restrict__ x, const float* __restrict__ stats,
                         const float* __restrict__ sc, const float* __restrict__ bi,
                         u16* __restrict__ h) {
    size_t idx = (size_t)blockIdx.x * 256 + threadIdx.x;   // [b][c][vox] flat
    int vox = (int)(idx & 4095);
    int c   = (int)((idx >> 12) & 511);
    int b   = (int)(idx >> 21);
    int bg  = b * 32 + (c >> 4);
    float v = (x[idx] - stats[bg * 2]) * stats[bg * 2 + 1] * sc[c] + bi[c];
    float y = v / (1.f + __expf(-v));
    int vd = (vox >> 8) + 1, vh = ((vox >> 4) & 15) + 1, vw = (vox & 15) + 1;
    h[((size_t)b * PVOL + vd * (PDIM*PDIM) + vh * PDIM + vw) * C_SZ + c] = f2bf(y);
}

// ---------------------------------------------------------------- implicit-GEMM 3x3x3 conv
// LDS-tiled: 8-wave block (2x4) computes a 64(M)x128(N) tile. Per tap and per
// 128-wide K-chunk the block cooperatively stages A (64x128, 16KB) and
// B (128x128, 32KB) into LDS, then each wave runs a 32x32 register tile
// (4 WMMA / K-step) fed by ds_load_b128. Cuts L2 traffic ~2.5x vs direct.
__global__ void __launch_bounds__(256)
conv3d_wmma(const u16* __restrict__ h, const u16* __restrict__ wt,
            const float* __restrict__ cb, u16* __restrict__ q) {
    __shared__ u16 As[64 * 128];    // 16 KB
    __shared__ u16 Bs[128 * 128];   // 32 KB

    int tnb = blockIdx.x & 3;       // 512/128 = 4 N blocks
    int tmb = blockIdx.x >> 2;      // 32768/64 = 512 M blocks
    int wave = threadIdx.x >> 5;
    int wn = wave & 3, wm = wave >> 2;
    int lane = threadIdx.x & 31, lh = lane & 15, half = lane >> 4;

    int m0 = tmb * 64;
    int b  = m0 >> 12;
    int vox_base = m0 & 4095;
    int n0 = tnb * 128;
    const u16* hb = h + (size_t)b * PVOL * C_SZ;

    // per-wave LDS fragment pointers
    const u16* as0 = As + (wm * 32 + lh) * 128 + half * 8;
    const u16* as1 = as0 + 16 * 128;
    const u16* bs0 = Bs + (wn * 32 + lh) * 128 + half * 8;
    const u16* bs1 = bs0 + 16 * 128;

    v8f acc[2][2];
    #pragma unroll
    for (int i = 0; i < 2; i++)
        #pragma unroll
        for (int j = 0; j < 2; j++) acc[i][j] = (v8f){0.f,0.f,0.f,0.f,0.f,0.f,0.f,0.f};

    for (int tap = 0; tap < 27; tap++) {
        int dd = tap / 9 - 1, dh = (tap / 3) % 3 - 1, dw = tap % 3 - 1;
        for (int k0 = 0; k0 < C_SZ; k0 += 128) {
            __syncthreads();              // previous chunk fully consumed
            // ---- stage A: 64 rows x 128 cin (4 x b128 per thread)
            #pragma unroll
            for (int j = 0; j < 4; j++) {
                int f  = threadIdx.x * 4 + j;
                int rr = f >> 4, c8 = (f & 15) * 8;
                int vox = vox_base + rr;
                int pd = (vox >> 8) + 1 + dd;
                int ph = ((vox >> 4) & 15) + 1 + dh;
                int pw = (vox & 15) + 1 + dw;
                const u16* g = hb + ((size_t)pd * (PDIM*PDIM) + ph * PDIM + pw) * C_SZ + k0 + c8;
                *(uint4*)(As + rr * 128 + c8) = *(const uint4*)g;
            }
            // ---- stage B: 128 couts x 128 cin (8 x b128 per thread)
            #pragma unroll
            for (int j = 0; j < 8; j++) {
                int f  = threadIdx.x * 8 + j;
                int cc = f >> 4, c8 = (f & 15) * 8;
                const u16* g = wt + ((size_t)tap * 512 + n0 + cc) * C_SZ + k0 + c8;
                *(uint4*)(Bs + cc * 128 + c8) = *(const uint4*)g;
            }
            __syncthreads();
            // ---- compute: 4 K-steps x 4 WMMA
            #pragma unroll
            for (int kk = 0; kk < 128; kk += 32) {
                FragAB A0, A1, B0, B1;
                A0.u[0] = *(const uint4*)(as0 + kk); A0.u[1] = *(const uint4*)(as0 + kk + 16);
                A1.u[0] = *(const uint4*)(as1 + kk); A1.u[1] = *(const uint4*)(as1 + kk + 16);
                B0.u[0] = *(const uint4*)(bs0 + kk); B0.u[1] = *(const uint4*)(bs0 + kk + 16);
                B1.u[0] = *(const uint4*)(bs1 + kk); B1.u[1] = *(const uint4*)(bs1 + kk + 16);
                acc[0][0] = wmma_bf16(A0.v, B0.v, acc[0][0]);
                acc[0][1] = wmma_bf16(A0.v, B1.v, acc[0][1]);
                acc[1][0] = wmma_bf16(A1.v, B0.v, acc[1][0]);
                acc[1][1] = wmma_bf16(A1.v, B1.v, acc[1][1]);
            }
        }
    }

    // epilogue: + conv bias + voxel PE -> q bf16 [b][vox][c]
    #pragma unroll
    for (int ns = 0; ns < 2; ns++) {
        int n = n0 + wn * 32 + ns * 16 + lh;
        float bias = cb[n];
        #pragma unroll
        for (int ms = 0; ms < 2; ms++)
            #pragma unroll
            for (int r = 0; r < 8; r++) {
                int mm = m0 + wm * 32 + ms * 16 + r + half * 8;
                q[(size_t)mm * C_SZ + n] = f2bf(acc[ms][ns][r] + bias + pe_val(mm & 4095, n));
            }
    }
}

// ---------------------------------------------------------------- generic bf16 WMMA GEMM
// out = A[M][K] * W[N][K]^T + bias; 32x32 tile per wave (4 WMMA / K-step)
#define MODE_PLAIN 0   // bf16 out[m*N+n]
#define MODE_KV1   1   // + cond PE(pos = m%196), bf16 out[m*512+n]
#define MODE_KH    2   // scatter to khT[b][h][s][dh]
#define MODE_VH    3   // scatter to vhT[b][h][dh][s]
#define MODE_OUT   4   // f32 transpose to out[b][c][l]
__global__ void gemm_bf16(const u16* __restrict__ A, const u16* __restrict__ W,
                          const float* __restrict__ bias, void* __restrict__ out,
                          int M, int N, int K, int mode) {
    int wid = (blockIdx.x * blockDim.x + threadIdx.x) >> 5;
    int ntiles = N >> 5;
    int tiles = (M >> 5) * ntiles;
    if (wid >= tiles) return;
    int tn = wid % ntiles, tm = wid / ntiles;
    int lane = threadIdx.x & 31, lh = lane & 15, half = lane >> 4;

    const u16* a0 = A + ((size_t)tm * 32 + lh) * K + half * 8;
    const u16* a1 = a0 + (size_t)16 * K;
    const u16* w0 = W + ((size_t)tn * 32 + lh) * K + half * 8;
    const u16* w1 = w0 + (size_t)16 * K;

    v8f acc[2][2];
    #pragma unroll
    for (int i = 0; i < 2; i++)
        #pragma unroll
        for (int j = 0; j < 2; j++) acc[i][j] = (v8f){0.f,0.f,0.f,0.f,0.f,0.f,0.f,0.f};

    #pragma unroll 2
    for (int k0 = 0; k0 < K; k0 += 32) {
        FragAB A0, A1, B0, B1;
        A0.u[0] = *(const uint4*)(a0 + k0); A0.u[1] = *(const uint4*)(a0 + k0 + 16);
        A1.u[0] = *(const uint4*)(a1 + k0); A1.u[1] = *(const uint4*)(a1 + k0 + 16);
        B0.u[0] = *(const uint4*)(w0 + k0); B0.u[1] = *(const uint4*)(w0 + k0 + 16);
        B1.u[0] = *(const uint4*)(w1 + k0); B1.u[1] = *(const uint4*)(w1 + k0 + 16);
        acc[0][0] = wmma_bf16(A0.v, B0.v, acc[0][0]);
        acc[0][1] = wmma_bf16(A0.v, B1.v, acc[0][1]);
        acc[1][0] = wmma_bf16(A1.v, B0.v, acc[1][0]);
        acc[1][1] = wmma_bf16(A1.v, B1.v, acc[1][1]);
    }

    #pragma unroll
    for (int ns = 0; ns < 2; ns++) {
        int n = tn * 32 + ns * 16 + lh;
        float bn = bias[n];
        #pragma unroll
        for (int ms = 0; ms < 2; ms++)
            #pragma unroll
            for (int r = 0; r < 8; r++) {
                int m = tm * 32 + ms * 16 + r + half * 8;
                float v = acc[ms][ns][r] + bn;
                if (mode == MODE_PLAIN) {
                    ((u16*)out)[(size_t)m * N + n] = f2bf(v);
                } else if (mode == MODE_KV1) {
                    ((u16*)out)[(size_t)m * N + n] = f2bf(v + pe_val(m % S_SZ, n));
                } else if (mode == MODE_KH) {
                    int s = m % S_SZ, bb = m / S_SZ, hh = n >> 6, dh = n & 63;
                    ((u16*)out)[(((size_t)(bb * HEADS + hh) * SPAD) + s) * HD + dh] = f2bf(v);
                } else if (mode == MODE_VH) {
                    int s = m % S_SZ, bb = m / S_SZ, hh = n >> 6, dh = n & 63;
                    ((u16*)out)[(((size_t)(bb * HEADS + hh) * HD) + dh) * SPAD + s] = f2bf(v);
                } else {
                    int bb = m >> 12, l = m & 4095;
                    ((float*)out)[((size_t)bb * C_SZ + n) * L_SZ + l] = v;
                }
            }
    }
}

// ---------------------------------------------------------------- flash attention
// per wave: 16 query rows for one (b,h). Online softmax over 7 S-blocks of 32.
__global__ void flash_attn(const u16* __restrict__ qh, const u16* __restrict__ khT,
                           const u16* __restrict__ vhT, const float* __restrict__ proj,
                           u16* __restrict__ o) {
    __shared__ u16 pbuf[8 * 16 * 32];
    int b = blockIdx.z, hd = blockIdx.y;
    int wave = threadIdx.x >> 5;
    int lane = threadIdx.x & 31, lh = lane & 15, half = lane >> 4;
    int l0 = (blockIdx.x * 8 + wave) * 16;

    FragAB aq0, aq1;
    {
        const u16* qb = qh + ((size_t)(b * L_SZ + l0 + lh)) * C_SZ + hd * HD + half * 8;
        aq0.u[0] = *(const uint4*)(qb);      aq0.u[1] = *(const uint4*)(qb + 16);
        aq1.u[0] = *(const uint4*)(qb + 32); aq1.u[1] = *(const uint4*)(qb + 48);
    }

    const float* P = proj + b * 9;
    float ps0[8], ps1[8];
    #pragma unroll
    for (int r = 0; r < 8; r++) {
        int vox = (l0 + r + half * 8) & 4095;
        float p0 = ((float)(vox >> 8)        + 0.5f) * (1.f / 16.f) - 0.5f;
        float p1 = ((float)((vox >> 4) & 15) + 0.5f) * (1.f / 16.f) - 0.5f;
        float p2 = ((float)(vox & 15)        + 0.5f) * (1.f / 16.f) - 0.5f;
        float c0 = P[0]*p0 + P[1]*p1 + P[2]*p2;
        float c1 = P[3]*p0 + P[4]*p1 + P[5]*p2;
        float c2 = P[6]*p0 + P[7]*p1 + P[8]*p2;
        float inv = 1.f / c2;
        float pix0 = fminf(fmaxf(rintf(c0 * inv), 0.f), 223.f);
        float pix1 = fminf(fmaxf(rintf(c1 * inv), 0.f), 223.f);
        ps0[r] = (2.f * pix1 + 1.f) * (14.f / 448.f) - 0.5f;
        ps1[r] = (2.f * (223.f - pix0) + 1.f) * (14.f / 448.f) - 0.5f;
    }

    float runmax[8], runsum[8];
    v8f ov[4];
    #pragma unroll
    for (int r = 0; r < 8; r++) { runmax[r] = -__builtin_inff(); runsum[r] = 0.f; }
    #pragma unroll
    for (int t = 0; t < 4; t++) ov[t] = (v8f){0.f,0.f,0.f,0.f,0.f,0.f,0.f,0.f};

    u16* pb = pbuf + wave * 512;
    const float msig2 = (1.5f - 1e-6f) * (1.5f - 1e-6f);
    const v8f zero8 = {0.f,0.f,0.f,0.f,0.f,0.f,0.f,0.f};

    for (int s0 = 0; s0 < SPAD; s0 += 32) {
        v8f sc[2];
        #pragma unroll
        for (int t = 0; t < 2; t++) {
            const u16* kb = khT + ((size_t)(b * HEADS + hd) * SPAD + s0 + t * 16 + lh) * HD + half * 8;
            FragAB b0, b1;
            b0.u[0] = *(const uint4*)(kb);      b0.u[1] = *(const uint4*)(kb + 16);
            b1.u[0] = *(const uint4*)(kb + 32); b1.u[1] = *(const uint4*)(kb + 48);
            v8f a = wmma_bf16(aq0.v, b0.v, zero8);
            sc[t]  = wmma_bf16(aq1.v, b1.v, a);
        }
        float mloc[8];
        #pragma unroll
        for (int r = 0; r < 8; r++) mloc[r] = -__builtin_inff();
        #pragma unroll
        for (int t = 0; t < 2; t++) {
            int scol = s0 + t * 16 + lh;
            float vi = (float)(scol / 14), vj = (float)(scol % 14);
            bool colpad = scol >= S_SZ;
            #pragma unroll
            for (int r = 0; r < 8; r++) {
                float di = ps0[r] - vi, dj = ps1[r] - vj;
                bool msk = colpad || (di * di + dj * dj > msig2);
                float v = msk ? -__builtin_inff() : sc[t][r] * 0.125f;
                sc[t][r] = v;
                mloc[r] = fmaxf(mloc[r], v);
            }
        }
        #pragma unroll
        for (int r = 0; r < 8; r++) {
            mloc[r] = fmaxf(mloc[r], __shfl_xor(mloc[r], 1, 32));
            mloc[r] = fmaxf(mloc[r], __shfl_xor(mloc[r], 2, 32));
            mloc[r] = fmaxf(mloc[r], __shfl_xor(mloc[r], 4, 32));
            mloc[r] = fmaxf(mloc[r], __shfl_xor(mloc[r], 8, 32));
        }
        float alpha[8];
        #pragma unroll
        for (int r = 0; r < 8; r++) {
            float nm = fmaxf(runmax[r], mloc[r]);
            alpha[r] = (nm == -__builtin_inff()) ? 1.f : __expf(runmax[r] - nm);
            runmax[r] = nm;
        }
        float sl[8];
        #pragma unroll
        for (int r = 0; r < 8; r++) sl[r] = 0.f;
        #pragma unroll
        for (int t = 0; t < 2; t++)
            #pragma unroll
            for (int r = 0; r < 8; r++) {
                float p = (sc[t][r] == -__builtin_inff()) ? 0.f : __expf(sc[t][r] - runmax[r]);
                sc[t][r] = p;
                sl[r] += p;
            }
        #pragma unroll
        for (int r = 0; r < 8; r++) {
            sl[r] += __shfl_xor(sl[r], 1, 32);
            sl[r] += __shfl_xor(sl[r], 2, 32);
            sl[r] += __shfl_xor(sl[r], 4, 32);
            sl[r] += __shfl_xor(sl[r], 8, 32);
            runsum[r] = runsum[r] * alpha[r] + sl[r];
        }
        #pragma unroll
        for (int t = 0; t < 4; t++)
            #pragma unroll
            for (int r = 0; r < 8; r++) ov[t][r] *= alpha[r];

        #pragma unroll
        for (int t = 0; t < 2; t++)
            #pragma unroll
            for (int r = 0; r < 8; r++)
                pb[(r + half * 8) * 32 + t * 16 + lh] = f2bf(sc[t][r]);
        asm volatile("s_wait_dscnt 0" ::: "memory");
        FragAB pa;
        {
            const u16* pr = pb + lh * 32 + half * 8;
            pa.u[0] = *(const uint4*)(pr);
            pa.u[1] = *(const uint4*)(pr + 16);
        }
        #pragma unroll
        for (int t = 0; t < 4; t++) {
            const u16* vb = vhT + ((size_t)(b * HEADS + hd) * HD + t * 16 + lh) * SPAD + s0 + half * 8;
            FragAB bv;
            bv.u[0] = *(const uint4*)(vb);
            bv.u[1] = *(const uint4*)(vb + 16);
            ov[t] = wmma_bf16(pa.v, bv.v, ov[t]);
        }
    }

    #pragma unroll
    for (int r = 0; r < 8; r++) {
        float inv = runsum[r] > 0.f ? 1.f / runsum[r] : 0.f;
        int m = l0 + r + half * 8;
        #pragma unroll
        for (int t = 0; t < 4; t++)
            o[((size_t)(b * L_SZ + m)) * C_SZ + hd * HD + t * 16 + lh] = f2bf(ov[t][r] * inv);
    }
}

// ---------------------------------------------------------------- host launcher
extern "C" void kernel_launch(void* const* d_in, const int* in_sizes, int n_in,
                              void* d_out, int out_size, void* d_ws, size_t ws_size,
                              hipStream_t stream) {
    const float* x      = (const float*)d_in[0];
    const float* sketch = (const float*)d_in[1];
    const float* proj   = (const float*)d_in[2];
    const float* gsc    = (const float*)d_in[3];
    const float* gbi    = (const float*)d_in[4];
    const float* convw  = (const float*)d_in[5];
    const float* convb  = (const float*)d_in[6];
    const float* kw     = (const float*)d_in[7];
    const float* kb     = (const float*)d_in[8];
    const float* vw     = (const float*)d_in[9];
    const float* vb     = (const float*)d_in[10];
    const float* ipw    = (const float*)d_in[11];
    const float* ipb    = (const float*)d_in[12];
    const float* ow     = (const float*)d_in[13];
    const float* ob     = (const float*)d_in[14];

    char* ws = (char*)d_ws;
    u16*   h     = (u16*)(ws + 0);              // 45.6 MB padded, later reused as qh
    u16*   q     = (u16*)(ws + 47775744);       // 33.5 MB, later reused as o
    u16*   wt    = (u16*)(ws + 81330176);       // 14.2 MB  conv weights [tap][co][ci]
    u16*   wproj = (u16*)(ws + 95485952);       // in_proj_w bf16 [1536][512]
    u16*   kwb   = (u16*)(ws + 97058816);
    u16*   vwb   = (u16*)(ws + 97845248);
    u16*   owb   = (u16*)(ws + 98631680);
    u16*   skb   = (u16*)(ws + 99155968);       // sketch bf16 [1568][768]
    u16*   k1    = (u16*)(ws + 101564416);      // [1568][512]
    u16*   v1    = (u16*)(ws + 103170048);
    u16*   khT   = (u16*)(ws + 104775680);      // [8][8][224][64]
    u16*   vhT   = (u16*)(ws + 106610688);      // [8][8][64][224]
    float* stats = (float*)(ws + 108445696);
    u16*   qh = h;                              // alias after conv consumes h
    u16*   o  = q;                              // alias after qh GEMM consumes q
    (void)in_sizes; (void)n_in; (void)out_size; (void)ws_size;

    // zero halo-padded h and the padded S regions of K^T / V^T
    fill_zero_u16<<<(23887872 + 255) / 256, 256, 0, stream>>>(h, 23887872);
    fill_zero_u16<<<(917504 + 255) / 256, 256, 0, stream>>>(khT, 917504);
    fill_zero_u16<<<(917504 + 255) / 256, 256, 0, stream>>>(vhT, 917504);

    // bf16 conversions
    cvt_bf16<<<(786432 + 255) / 256, 256, 0, stream>>>(ipw, wproj, 786432);
    cvt_bf16<<<(393216 + 255) / 256, 256, 0, stream>>>(kw, kwb, 393216);
    cvt_bf16<<<(393216 + 255) / 256, 256, 0, stream>>>(vw, vwb, 393216);
    cvt_bf16<<<(262144 + 255) / 256, 256, 0, stream>>>(ow, owb, 262144);
    cvt_bf16<<<(1204224 + 255) / 256, 256, 0, stream>>>(sketch, skb, 1204224);
    permute_convw<<<(7077888 + 255) / 256, 256, 0, stream>>>(convw, wt);

    // GroupNorm + SiLU -> padded h (bf16, channels-last)
    gn_stats<<<256, 256, 0, stream>>>(x, stats);
    gn_apply<<<65536, 256, 0, stream>>>(x, stats, gsc, gbi, h);

    // conv3d (implicit GEMM, LDS-tiled 64x128 block tile) + bias + voxel PE -> q
    conv3d_wmma<<<2048, 256, 0, stream>>>(h, wt, convb, q);

    // k = sketch@k_w^T + k_b + PE ; v likewise
    gemm_bf16<<<98, 256, 0, stream>>>(skb, kwb, kb, k1, 1568, 512, 768, MODE_KV1);
    gemm_bf16<<<98, 256, 0, stream>>>(skb, vwb, vb, v1, 1568, 512, 768, MODE_KV1);
    // kh = k@wk^T + bk  (scatter to [b][h][s][dh]) ; vh -> [b][h][dh][s]
    gemm_bf16<<<98, 256, 0, stream>>>(k1, wproj + 512 * 512,     ipb + 512,  khT, 1568, 512, 512, MODE_KH);
    gemm_bf16<<<98, 256, 0, stream>>>(v1, wproj + 2 * 512 * 512, ipb + 1024, vhT, 1568, 512, 512, MODE_VH);
    // qh = q@wq^T + bq
    gemm_bf16<<<2048, 256, 0, stream>>>(q, wproj, ipb, qh, 32768, 512, 512, MODE_PLAIN);

    // flash attention -> o
    flash_attn<<<dim3(32, HEADS, B_SZ), 256, 0, stream>>>(qh, khT, vhT, proj, o);

    // out = o@out_w^T + out_b, transposed to [b][c][l] f32
    gemm_bf16<<<2048, 256, 0, stream>>>(o, owb, ob, d_out, 32768, 512, 512, MODE_OUT);
}
